// DVTATT_88158498718224
// MI455X (gfx1250) — matrix operand order repaired and checked by reading, not compiled, not run on hardware
//
#include <hip/hip_runtime.h>
#include <hip/hip_bf16.h>

// ---------------------------------------------------------------------------
// MI455X (gfx1250, wave32) implementation of the masked cross-attention block.
// All heavy math runs through v_wmma_f32_16x16x32_bf16 (f32 accumulate).
// Shapes: b=4, c=256, n=4096 (64x64), conv-in=1024, fuse-in=512.
// GEMM loops use fully-unrolled ping-pong buffers: next chunk's
// global_load_b128 clause issues before the current chunk's WMMA chain, with
// compile-time buffer indices so no register-rotation copies are generated.
// ---------------------------------------------------------------------------

typedef __bf16 v16bf __attribute__((ext_vector_type(16)));
typedef float  v8f   __attribute__((ext_vector_type(8)));
typedef unsigned int v8u __attribute__((ext_vector_type(8)));

#define WMMA_BF16(A, B, C) \
  __builtin_amdgcn_wmma_f32_16x16x32_bf16(false, (A), false, (B), (short)0, (C), false, false)

__device__ __forceinline__ unsigned short f2bf(float f) {
  unsigned int u = __float_as_uint(f);
  u += 0x7FFFu + ((u >> 16) & 1u);          // round-to-nearest-even
  return (unsigned short)(u >> 16);
}
__device__ __forceinline__ unsigned int pack2bf(float lo, float hi) {
  return (unsigned int)f2bf(lo) | ((unsigned int)f2bf(hi) << 16);
}

// A-fragment (16x32, bf16, M x K). src is row-major [row][K], bf16.
// Layout: lanes 0-15 row M=lane, K 0..15 (VGPR r -> K=2r,2r+1);
//         lanes 16-31 row M=lane-16, K 16..31.
// Per lane: one contiguous 32B run -> compiler emits global_load_b128 x2.
__device__ __forceinline__ v16bf load_frag_A(const unsigned short* src, int stride,
                                             int rowbase, int kbase, int lane) {
  const int row  = rowbase + (lane & 15);
  const int koff = (lane & 16) ? 16 : 0;
  v8u u;
#pragma unroll
  for (int r = 0; r < 8; ++r) {
    int k = kbase + koff + 2 * r;
    u[r] = *(const unsigned int*)(src + (size_t)row * stride + k);
  }
  return __builtin_bit_cast(v16bf, u);
}

// B-fragment (32x16, bf16, K x N). src is row-major [N][K], bf16.
// Layout (contiguous k, same as load_frag_A): lanes 0-15 hold K=0..15 (VGPR r -> K=2r,2r+1), N=lane;
//             lanes 16-31 hold K=16..31, N=lane-16.
__device__ __forceinline__ v16bf load_frag_B(const unsigned short* src, int stride,
                                             int nbase, int kbase, int lane) {
  const int n    = nbase + (lane & 15);
  const int koff = (lane & 16) ? 16 : 0;
  v8u u;
#pragma unroll
  for (int r = 0; r < 8; ++r) {
    int k = kbase + koff + 2 * r;
    u[r] = *(const unsigned int*)(src + (size_t)n * stride + k);
  }
  return __builtin_bit_cast(v16bf, u);
}
// C/D fragment (16x16 f32): VGPR r, lane l -> row M = r + 8*(l>=16), col N = l&15.

// ---------------------------------------------------------------------------
// 1) maxpool(8,8) of mask -> binary m[b][4096]
__global__ void k_maskpool(const float* __restrict__ mask, float* __restrict__ m) {
  int i = blockIdx.x * 256 + threadIdx.x;      // b*4096 + p
  int b = i >> 12, p = i & 4095;
  int y = p >> 6, x = p & 63;
  const float* s = mask + ((size_t)b * 512 + y * 8) * 512 + x * 8;
  float mx = 0.f;
#pragma unroll
  for (int dy = 0; dy < 8; ++dy)
#pragma unroll
    for (int dx = 0; dx < 8; ++dx) mx = fmaxf(mx, s[dy * 512 + dx]);
  m[i] = (mx > 0.f) ? 1.f : 0.f;
}

// 2) convert K_w (256x1024) and fuse_w (256x512) to bf16
__global__ void k_cvt_weights(const float* __restrict__ Kw, const float* __restrict__ Fw,
                              unsigned short* __restrict__ Kwbf, unsigned short* __restrict__ Fwbf) {
  int i = blockIdx.x * 256 + threadIdx.x;
  if (i < 262144)       Kwbf[i] = f2bf(Kw[i]);
  else                  Fwbf[i - 262144] = f2bf(Fw[i - 262144]);
}

// 3) build inT[b][p][1024] bf16 = transpose of concat(x, dvt).
//    One thread per channel-PAIR: coalesced f32 reads, packed dword stores.
__global__ void k_transpose_in(const float* __restrict__ x, const float* __restrict__ dvt,
                               unsigned short* __restrict__ inT) {
  size_t i = (size_t)blockIdx.x * 256 + threadIdx.x;  // (b*512 + cp)*4096 + p
  int p  = (int)(i & 4095);
  size_t t = i >> 12;
  int cp = (int)(t & 511);
  int b  = (int)(t >> 9);
  int ch = cp * 2;
  float v0, v1;
  if (ch < 256) {
    const float* s = x + ((size_t)b * 256 + ch) * 4096 + p;
    v0 = s[0]; v1 = s[4096];
  } else {
    const float* s = dvt + ((size_t)b * 768 + (ch - 256)) * 4096 + p;
    v0 = s[0]; v1 = s[4096];
  }
  *(unsigned int*)(inT + ((size_t)b * 4096 + p) * 1024 + ch) = pack2bf(v0, v1);
}

// 4) featsT[b][p][256] = inT[b][p][:] * K_w^T + K_b
//    WMMA GEMM, 4 output tiles per wave, fully-unrolled ping-pong K-loop.
__global__ void __launch_bounds__(256) k_feats_gemm(const unsigned short* __restrict__ inT,
                                                    const unsigned short* __restrict__ Kwbf,
                                                    const float* __restrict__ Kb,
                                                    float* __restrict__ featsT) {
  int lane = threadIdx.x & 31, wid = threadIdx.x >> 5;
  int t = blockIdx.x * 8 + wid;                // wave-tile id, 4096 total
  int b = t >> 10;
  int rem = t & 1023;
  int ptile = rem >> 2, og = rem & 3;          // og: group of 4 otiles
  const unsigned short* A = inT + (size_t)b * 4096 * 1024;
  v8f d[4];
#pragma unroll
  for (int j = 0; j < 4; ++j) d[j] = (v8f){};

  v16bf a[2];
  v16bf bb[2][4];
  a[0] = load_frag_A(A, 1024, ptile * 16, 0, lane);
#pragma unroll
  for (int j = 0; j < 4; ++j) bb[0][j] = load_frag_B(Kwbf, 1024, (og * 4 + j) * 16, 0, lane);

#pragma unroll
  for (int c = 0; c < 32; ++c) {               // 32 chunks of K=32
    const int cur = c & 1, nxt = cur ^ 1;
    if (c + 1 < 32) {                          // next chunk's loads go in flight
      a[nxt] = load_frag_A(A, 1024, ptile * 16, (c + 1) * 32, lane);
#pragma unroll
      for (int j = 0; j < 4; ++j)
        bb[nxt][j] = load_frag_B(Kwbf, 1024, (og * 4 + j) * 16, (c + 1) * 32, lane);
    }
#pragma unroll
    for (int j = 0; j < 4; ++j) d[j] = WMMA_BF16(a[cur], bb[cur][j], d[j]);
  }

  int h = (lane >> 4) & 1;
  float* dst = featsT + (size_t)b * 4096 * 256;
#pragma unroll
  for (int j = 0; j < 4; ++j) {
    int o = (og * 4 + j) * 16 + (lane & 15);
    float bias = Kb[o];
#pragma unroll
    for (int r = 0; r < 8; ++r)
      dst[(size_t)(ptile * 16 + r + 8 * h) * 256 + o] = d[j][r] + bias;
  }
}

// 5) per-pixel L2 normalize over c=256 -> Nbf[b][p][256] bf16 (one wave per pixel)
__global__ void k_normalize(const float* __restrict__ featsT, unsigned short* __restrict__ Nbf) {
  int lane = threadIdx.x & 31, wid = threadIdx.x >> 5;
  int pix = blockIdx.x * 8 + wid;              // 0..16383
  const float* fr = featsT + (size_t)pix * 256;
  float v[8], ss = 0.f;
#pragma unroll
  for (int j = 0; j < 8; ++j) { v[j] = fr[lane + 32 * j]; ss += v[j] * v[j]; }
#pragma unroll
  for (int off = 16; off > 0; off >>= 1) ss += __shfl_xor(ss, off, 32);
  float inv = 1.f / (sqrtf(ss) + 1e-8f);
  unsigned short* nr = Nbf + (size_t)pix * 256;
#pragma unroll
  for (int j = 0; j < 8; ++j) nr[lane + 32 * j] = f2bf(v[j] * inv);
}

// 6) Vbf[b][c][p] = bf16( x * (1-m) )
__global__ void k_make_v(const float* __restrict__ x, const float* __restrict__ m,
                         unsigned short* __restrict__ Vbf) {
  size_t i = (size_t)blockIdx.x * 256 + threadIdx.x;   // (b*256+c)*4096 + p
  int p = (int)(i & 4095);
  int b = (int)(i >> 20);                              // 256*4096 = 2^20 per batch
  float bm = 1.f - m[b * 4096 + p];
  Vbf[i] = f2bf(x[i] * bm);
}

// 7) flash attention: per wave 16 query rows, fixed softmax max = 20
//    (valid since Q,K unit-normalized => s in [-1,1], 20*s <= 20: no overflow).
//    pass1: denom; pass2: att tiles -> (att x V) via LDS repack, + visatt colsums.
__global__ void __launch_bounds__(256) k_attention(const unsigned short* __restrict__ Nbf,
                                                   const unsigned short* __restrict__ Vbf,
                                                   const float* __restrict__ mF,
                                                   float* __restrict__ visatt,
                                                   unsigned short* __restrict__ attforeT) {
  __shared__ float plds[8][32][16];            // per-wave 32k x 16q att staging
  int lane = threadIdx.x & 31, wid = threadIdx.x >> 5;
  int b = blockIdx.x >> 5;                     // 32 blocks per batch
  int qchunk = blockIdx.x & 31;
  int qbase = qchunk * 128 + wid * 16;
  const unsigned short* Nb = Nbf + (size_t)b * 4096 * 256;
  const unsigned short* Vb = Vbf + (size_t)b * 256 * 4096;
  const float* mb = mF + b * 4096;
  int half = (lane >> 4) & 1;
  int ln = lane & 15;

  v16bf qf[8];
#pragma unroll
  for (int cb = 0; cb < 8; ++cb) qf[cb] = load_frag_A(Nb, 256, qbase, cb * 32, lane);

  // ---- pass 1: softmax denominators (p = km * exp(20(s-1))) ----
  float acc[8];
#pragma unroll
  for (int r = 0; r < 8; ++r) acc[r] = 0.f;
#pragma unroll 2
  for (int kb = 0; kb < 4096; kb += 16) {
    // prefetch next key tile (global_prefetch_b8) while WMMAs run
    __builtin_prefetch(Nb + (size_t)(((kb + 16) & 4095) + ln) * 256, 0, 0);
    v16bf kf[8];                               // batch-load: 16 b128 loads in flight
#pragma unroll
    for (int cb = 0; cb < 8; ++cb) kf[cb] = load_frag_B(Nb, 256, kb, cb * 32, lane);
    v8f s = {};
#pragma unroll
    for (int cb = 0; cb < 8; ++cb) s = WMMA_BF16(qf[cb], kf[cb], s);
    float km = 1.f - mb[kb + ln];
#pragma unroll
    for (int r = 0; r < 8; ++r) acc[r] += km * __expf(20.f * s[r] - 20.f);
  }
#pragma unroll
  for (int r = 0; r < 8; ++r) {
    float a = acc[r];
    a += __shfl_xor(a, 1, 32); a += __shfl_xor(a, 2, 32);
    a += __shfl_xor(a, 4, 32); a += __shfl_xor(a, 8, 32);
    acc[r] = a;
  }
  float invd[8];                                // qm folded in: zeroes invalid rows
#pragma unroll
  for (int r = 0; r < 8; ++r)
    invd[r] = mb[qbase + r + 8 * half] / (acc[r] + 1e-30f);

  // ---- pass 2: att tiles, colsums (visatt), O^T = V x att^T ----
  v8f o[16];
#pragma unroll
  for (int cc = 0; cc < 16; ++cc) o[cc] = (v8f){};
  float* L = &plds[wid][0][0];
  for (int kb = 0; kb < 4096; kb += 32) {
#pragma unroll
    for (int sub = 0; sub < 2; ++sub) {
      int kk = kb + sub * 16;
      __builtin_prefetch(Nb + (size_t)(((kk + 32) & 4095) + ln) * 256, 0, 0);
      v8f s = {};
#pragma unroll
      for (int g = 0; g < 2; ++g) {            // 2 groups of 4 to bound VGPRs
        v16bf kf[4];
#pragma unroll
        for (int j = 0; j < 4; ++j)
          kf[j] = load_frag_B(Nb, 256, kk, (g * 4 + j) * 32, lane);
#pragma unroll
        for (int j = 0; j < 4; ++j) s = WMMA_BF16(qf[g * 4 + j], kf[j], s);
      }
      float km = 1.f - mb[kk + ln];
      float csum = 0.f;
#pragma unroll
      for (int r = 0; r < 8; ++r) {
        float att = invd[r] * km * __expf(20.f * s[r] - 20.f);
        L[(sub * 16 + ln) * 16 + r + 8 * half] = att;   // store [k_local][q_local]
        csum += att;
      }
      csum += __shfl_xor(csum, 16, 32);
      if (lane < 16) atomicAdd(&visatt[b * 4096 + kk + lane], csum);
    }
    __syncthreads();
    v8u bu;                                    // B-fragment of att^T [32k x 16q]
#pragma unroll
    for (int r = 0; r < 8; ++r) {
      int k = 2 * r + (half ? 16 : 0);
      bu[r] = pack2bf(L[k * 16 + ln], L[(k + 1) * 16 + ln]);
    }
    v16bf attB = __builtin_bit_cast(v16bf, bu);
    __syncthreads();
#pragma unroll
    for (int g = 0; g < 4; ++g) {              // A = V [16c x 32k], 4 frags in flight
      v16bf vf[4];
#pragma unroll
      for (int j = 0; j < 4; ++j)
        vf[j] = load_frag_A(Vb, 4096, (g * 4 + j) * 16, kb, lane);
#pragma unroll
      for (int j = 0; j < 4; ++j) o[g * 4 + j] = WMMA_BF16(vf[j], attB, o[g * 4 + j]);
    }
  }
  // store att_fore^T[b][q][c] bf16 (D: row=c, col=q), packed dword stores
  unsigned int* dst32 = (unsigned int*)(attforeT + (size_t)b * 4096 * 256 +
                                        (size_t)(qbase + ln) * 256);
#pragma unroll
  for (int cc = 0; cc < 16; ++cc)
#pragma unroll
    for (int r = 0; r < 8; r += 2)
      dst32[(cc * 16 + 8 * half + r) >> 1] = pack2bf(o[cc][r], o[cc][r + 1]);
}

// 8) fusedT[b][p][256] = concat(att_fore, x)[p][:] * fuse_w^T
//    WMMA GEMM, 4 output tiles per wave, fully-unrolled ping-pong K-loop.
__global__ void __launch_bounds__(256) k_fuse_gemm(const unsigned short* __restrict__ attforeT,
                                                   const unsigned short* __restrict__ inT,
                                                   const unsigned short* __restrict__ Fwbf,
                                                   float* __restrict__ fusedT) {
  int lane = threadIdx.x & 31, wid = threadIdx.x >> 5;
  int t = blockIdx.x * 8 + wid;                // 4096 wave-tiles
  int b = t >> 10;
  int rem = t & 1023;
  int ptile = rem >> 2, og = rem & 3;
  const unsigned short* A1 = attforeT + (size_t)b * 4096 * 256;  // i in [0,256)
  const unsigned short* A2 = inT + (size_t)b * 4096 * 1024;      // x = cols [0,256)
  auto loadA = [&](int ib) {
    return (ib < 256) ? load_frag_A(A1, 256, ptile * 16, ib, lane)
                      : load_frag_A(A2, 1024, ptile * 16, ib - 256, lane);
  };
  v8f d[4];
#pragma unroll
  for (int j = 0; j < 4; ++j) d[j] = (v8f){};

  v16bf a[2];
  v16bf bb[2][4];
  a[0] = loadA(0);
#pragma unroll
  for (int j = 0; j < 4; ++j) bb[0][j] = load_frag_B(Fwbf, 512, (og * 4 + j) * 16, 0, lane);

#pragma unroll
  for (int c = 0; c < 16; ++c) {               // 16 chunks of K=32
    const int cur = c & 1, nxt = cur ^ 1;
    if (c + 1 < 16) {
      a[nxt] = loadA((c + 1) * 32);
#pragma unroll
      for (int j = 0; j < 4; ++j)
        bb[nxt][j] = load_frag_B(Fwbf, 512, (og * 4 + j) * 16, (c + 1) * 32, lane);
    }
#pragma unroll
    for (int j = 0; j < 4; ++j) d[j] = WMMA_BF16(a[cur], bb[cur][j], d[j]);
  }

  int h = (lane >> 4) & 1;
  float* dst = fusedT + (size_t)b * 4096 * 256;
#pragma unroll
  for (int j = 0; j < 4; ++j) {
    int o = (og * 4 + j) * 16 + (lane & 15);
#pragma unroll
    for (int r = 0; r < 8; ++r)
      dst[(size_t)(ptile * 16 + r + 8 * h) * 256 + o] = d[j][r];
  }
}

// 9) out[b][c][p] = m ? fused : x
__global__ void k_compose(const float* __restrict__ fusedT, const float* __restrict__ x,
                          const float* __restrict__ m, float* __restrict__ out) {
  size_t i = (size_t)blockIdx.x * 256 + threadIdx.x;   // (b*256+o)*4096 + p
  int p = (int)(i & 4095);
  size_t t = i >> 12;
  int o = (int)(t & 255);
  int b = (int)(t >> 8);
  float mv = m[b * 4096 + p];
  out[i] = (mv > 0.5f) ? fusedT[((size_t)(b * 4096 + p)) * 256 + o] : x[i];
}

// 10) global max of visatt (16384 values, one workgroup)
__global__ void k_reduce_max(const float* __restrict__ visatt, float* __restrict__ maxv) {
  __shared__ float sm[256];
  float mx = 0.f;
  for (int i = threadIdx.x; i < 16384; i += 256) mx = fmaxf(mx, visatt[i]);
  sm[threadIdx.x] = mx;
  __syncthreads();
  for (int s = 128; s > 0; s >>= 1) {
    if (threadIdx.x < (unsigned)s) sm[threadIdx.x] = fmaxf(sm[threadIdx.x], sm[threadIdx.x + s]);
    __syncthreads();
  }
  if (threadIdx.x == 0) maxv[0] = sm[0];
}

// 11) attmask[b][512][512] = upsample8(visatt)/max
__global__ void k_attmask(const float* __restrict__ visatt, const float* __restrict__ maxv,
                          float* __restrict__ out2) {
  int i = blockIdx.x * 256 + threadIdx.x;      // b*262144 + Y*512 + X
  int X = i & 511, Y = (i >> 9) & 511, b = i >> 18;
  float inv = 1.f / maxv[0];
  out2[i] = visatt[b * 4096 + (Y >> 3) * 64 + (X >> 3)] * inv;
}

// ---------------------------------------------------------------------------
extern "C" void kernel_launch(void* const* d_in, const int* in_sizes, int n_in,
                              void* d_out, int out_size, void* d_ws, size_t ws_size,
                              hipStream_t stream) {
  (void)in_sizes; (void)n_in; (void)out_size; (void)ws_size;
  const float* x    = (const float*)d_in[0];   // [4,256,64,64]
  const float* mask = (const float*)d_in[1];   // [4,1,512,512]
  const float* dvt  = (const float*)d_in[2];   // [4,768,64,64]
  const float* Kw   = (const float*)d_in[3];   // [256,1024]
  const float* Kb   = (const float*)d_in[4];   // [256]
  const float* Fw   = (const float*)d_in[5];   // [256,512]
  float* out  = (float*)d_out;                 // 4,194,304 floats (out)
  float* out2 = out + 4194304;                 // 1,048,576 floats (attmask)

  char* w = (char*)d_ws;                       // workspace layout (256B aligned)
  float*          m        = (float*)(w + 0);                 //    65,536 B
  float*          visatt   = (float*)(w + 65536);             //    65,536 B
  float*          maxv     = (float*)(w + 131072);            //       256 B
  unsigned short* Kwbf     = (unsigned short*)(w + 131328);   //   524,288 B
  unsigned short* Fwbf     = (unsigned short*)(w + 655616);   //   262,144 B
  unsigned short* inT      = (unsigned short*)(w + 917760);   // 33,554,432 B
  float*          featsT   = (float*)(w + 34472192);          // 16,777,216 B
  unsigned short* Nbf      = (unsigned short*)(w + 51249408); //  8,388,608 B
  unsigned short* Vbf      = (unsigned short*)(w + 59638016); //  8,388,608 B
  unsigned short* attforeT = (unsigned short*)(w + 68026624); //  8,388,608 B
  float*          fusedT   = (float*)(w + 76415232);          // 16,777,216 B  (~93 MB total)

  hipMemsetAsync(w + 65536, 0, 65536 + 256, stream);          // visatt + maxv

  k_cvt_weights <<<1536,  256, 0, stream>>>(Kw, Fw, Kwbf, Fwbf);
  k_maskpool    <<<64,    256, 0, stream>>>(mask, m);
  k_transpose_in<<<32768, 256, 0, stream>>>(x, dvt, inT);
  k_feats_gemm  <<<512,   256, 0, stream>>>(inT, Kwbf, Kb, featsT);
  k_normalize   <<<2048,  256, 0, stream>>>(featsT, Nbf);
  k_make_v      <<<16384, 256, 0, stream>>>(x, m, Vbf);
  k_attention   <<<128,   256, 0, stream>>>(Nbf, Vbf, m, visatt, attforeT);
  k_fuse_gemm   <<<512,   256, 0, stream>>>(attforeT, inT, Fwbf, fusedT);
  k_compose     <<<16384, 256, 0, stream>>>(fusedT, x, m, out);
  k_reduce_max  <<<1,     256, 0, stream>>>(visatt, maxv);
  k_attmask     <<<4096,  256, 0, stream>>>(visatt, maxv, out2);
}
